// MultiHeadAttention_87359634801304
// MI455X (gfx1250) — compile-verified, matrix operand order checked by reading, not run on hardware
//
#include <hip/hip_runtime.h>
#include <cstdint>

#define D_DIM 1024
#define T_LEN 512
#define N_B   32
#define N_H   16
#define H_D   64
#define B_T   (N_B * T_LEN)   // 16384

typedef __attribute__((ext_vector_type(16))) __bf16 bf16x16;
typedef __attribute__((ext_vector_type(8)))  float  f32x8;

union Frag  { uint4 u[2]; bf16x16 v; };
union Pack8 { unsigned short s[8]; uint4 u; };

__device__ __forceinline__ unsigned short f2bf(float f) {
    unsigned u = __builtin_bit_cast(unsigned, f);
    u = (u + 0x7fffu + ((u >> 16) & 1u)) >> 16;   // round-to-nearest-even
    return (unsigned short)u;
}

__device__ __forceinline__ f32x8 wmma_bf16(const Frag& a, const Frag& b, f32x8 c) {
    return __builtin_amdgcn_wmma_f32_16x16x32_bf16(false, a.v, false, b.v,
                                                   (short)0, c, false, false);
}

// ---- CDNA5 async global->LDS DMA (ASYNCcnt-tracked), 16B per lane ----------
__device__ __forceinline__ void async_load_b128(unsigned lds_byte_addr,
                                                const void* gaddr) {
    asm volatile("global_load_async_to_lds_b128 %0, %1, off"
                 :
                 : "v"(lds_byte_addr), "v"((unsigned long long)(uintptr_t)gaddr)
                 : "memory");
}
__device__ __forceinline__ void wait_async_le6() {
    asm volatile("s_wait_asynccnt 0x6" ::: "memory");
}
__device__ __forceinline__ void wait_async_0() {
    asm volatile("s_wait_asynccnt 0x0" ::: "memory");
}

// ---------------------------------------------------------------- conversions
__global__ void k_f32_to_bf16(const float* __restrict__ in,
                              unsigned short* __restrict__ out, int n) {
    int i = blockIdx.x * blockDim.x + threadIdx.x;
    if (i < n) out[i] = f2bf(in[i]);
}

// in: [batch][K][N] fp32  ->  out: [batch][N][K] bf16
__global__ void k_f32_to_bf16_T(const float* __restrict__ in,
                                unsigned short* __restrict__ out,
                                int K, int N, int total) {
    int i = blockIdx.x * blockDim.x + threadIdx.x;
    if (i >= total) return;
    int z = i / (K * N);
    int r = i % (K * N);
    int n = r / K;
    int k = r % K;
    out[(size_t)z * N * K + (size_t)n * K + k] =
        f2bf(in[(size_t)z * K * N + (size_t)k * N + n]);
}

// ------------------------------------------------------------------ WMMA GEMM
// C[M,N] = A[M,K](bf16 rowmajor) @ Bt[N,K](bf16 rowmajor)^T + bias
// 128 threads = 4 waves; 128x64 block tile; each wave computes 32x64
// (waves stacked along M, B tile shared by all 4 waves).
// A/B tiles staged in LDS via global_load_async_to_lds_b128, double-buffered.
// MODE 0: bf16 C row-major; MODE 1: bf16 C^T into [B][H_D][T_LEN]; MODE 2: f32 C.
template <int MODE>
__global__ __launch_bounds__(128)
void k_gemm_wmma(const unsigned short* __restrict__ A,
                 const unsigned short* __restrict__ Bt,
                 const float* __restrict__ bias,
                 void* __restrict__ Cout,
                 int M, int N, int K,
                 long long aBatch, long long btBatch,
                 long long biasBatch, long long cBatch) {
    __shared__ __align__(16) unsigned short sA[2][128 * 32];  // 16 KB
    __shared__ __align__(16) unsigned short sB[2][64 * 32];   //  8 KB

    const int tid  = threadIdx.x;
    const int lane = tid & 31;
    const int wave = tid >> 5;           // 0..3 -> M offset wave*32
    const int cN   = lane & 15;
    const int ch   = lane >> 4;
    const int n0b  = blockIdx.x * 64;
    const int m0b  = blockIdx.y * 128;
    const int z    = blockIdx.z;
    (void)M;
    A    += (size_t)z * aBatch;
    Bt   += (size_t)z * btBatch;
    bias += (size_t)z * biasBatch;

    // stage one 128x32 A tile + 64x32 B tile into buffer `buf` (6 async/thread)
    auto stage = [&](int buf, int k0) {
        const unsigned aoff = (unsigned)(uintptr_t)&sA[buf][0];
        const unsigned boff = (unsigned)(uintptr_t)&sB[buf][0];
#pragma unroll
        for (int j = 0; j < 4; ++j) {             // A: 512 x 16B chunks
            const int cc  = j * 128 + tid;
            const int row = cc >> 2;
            const int c8  = (cc & 3) * 8;
            async_load_b128(aoff + cc * 16,
                            A + (size_t)(m0b + row) * K + k0 + c8);
        }
#pragma unroll
        for (int j = 0; j < 2; ++j) {             // B: 256 x 16B chunks
            const int cc  = j * 128 + tid;
            const int row = cc >> 2;
            const int c8  = (cc & 3) * 8;
            async_load_b128(boff + cc * 16,
                            Bt + (size_t)(n0b + row) * K + k0 + c8);
        }
    };

    f32x8 acc[2][4];
#pragma unroll
    for (int mt = 0; mt < 2; ++mt)
#pragma unroll
        for (int nt = 0; nt < 4; ++nt) acc[mt][nt] = (f32x8){};

    stage(0, 0);
    if (K > 32) stage(1, 32);

    for (int k0 = 0; k0 < K; k0 += 32) {
        const int cur = (k0 >> 5) & 1;
        // current buffer complete (async loads finish in order per wave)
        if (k0 + 32 < K) wait_async_le6(); else wait_async_0();
        __syncthreads();

        Frag a[2], b[4];
#pragma unroll
        for (int mt = 0; mt < 2; ++mt) {
            const uint4* p = (const uint4*)&sA[cur][(wave * 32 + mt * 16 + cN) * 32
                                                    + ch * 8];
            a[mt].u[0] = p[0];
            a[mt].u[1] = p[2];
        }
#pragma unroll
        for (int nt = 0; nt < 4; ++nt) {
            const uint4* p = (const uint4*)&sB[cur][(nt * 16 + cN) * 32 + ch * 8];
            b[nt].u[0] = p[0];
            b[nt].u[1] = p[2];
        }
        __syncthreads();                 // all waves done reading `cur`
        if (k0 + 64 < K) stage(cur, k0 + 64);

#pragma unroll
        for (int mt = 0; mt < 2; ++mt)
#pragma unroll
            for (int nt = 0; nt < 4; ++nt)
                acc[mt][nt] = wmma_bf16(a[mt], b[nt], acc[mt][nt]);
    }

#pragma unroll
    for (int nt = 0; nt < 4; ++nt) {
        const int n    = n0b + nt * 16 + cN;
        const float bv = bias[n];
#pragma unroll
        for (int mt = 0; mt < 2; ++mt) {
            f32x8 c = acc[mt][nt];
            const int mrow = m0b + wave * 32 + mt * 16 + ch * 8;
            if (MODE == 0) {
                unsigned short* C = (unsigned short*)Cout + (size_t)z * cBatch;
#pragma unroll
                for (int r = 0; r < 8; ++r)
                    C[(size_t)(mrow + r) * N + n] = f2bf(c[r] + bv);
            } else if (MODE == 2) {
                float* C = (float*)Cout + (size_t)z * cBatch;
#pragma unroll
                for (int r = 0; r < 8; ++r)
                    C[(size_t)(mrow + r) * N + n] = c[r] + bv;
            } else {  // MODE 1: transposed bf16 store into [B][H_D][T_LEN]
                int bb = mrow / T_LEN;            // 8 consecutive rows, same b
                int tt = mrow % T_LEN;
                Pack8 pk;
#pragma unroll
                for (int r = 0; r < 8; ++r) pk.s[r] = f2bf(c[r] + bv);
                unsigned short* C = (unsigned short*)Cout + (size_t)z * cBatch
                                  + ((size_t)bb * H_D + n) * T_LEN + tt;
                *(uint4*)C = pk.u;
            }
        }
    }
}

// ------------------------------------------------------------- attention core
// wei[t,s] = k[t]·q[s], causal (keep s<=t), softmax over s, out = wei @ v.
// One wave per (b, h, 16-row t-tile), streaming over s-blocks of 32.
// MASK=false for fully-valid blocks (s0+31 <= t0), MASK=true for the diagonal.
template <bool MASK>
__device__ __forceinline__ void attn_block(
    int s0, int t0, int cN, int ch,
    const unsigned short* __restrict__ qh,
    const unsigned short* __restrict__ vt,
    const Frag (&ak)[2], unsigned short* pl,
    float (&mrun)[8], float (&lrun)[8], f32x8 (&acc)[4]) {
    // ---- S = K_tile @ Q_block^T : two 16x16 f32 tiles ----
    f32x8 c[2];
#pragma unroll
    for (int nt = 0; nt < 2; ++nt) {
        Frag bq0, bq1;
        const size_t qrow = (size_t)(s0 + nt * 16 + cN) * H_D;
        const uint4* p0 = (const uint4*)(qh + qrow + 0  + ch * 8);
        const uint4* p1 = (const uint4*)(qh + qrow + 32 + ch * 8);
        bq0.u[0] = p0[0]; bq0.u[1] = p0[2];
        bq1.u[0] = p1[0]; bq1.u[1] = p1[2];
        f32x8 zz = (f32x8){};
        zz = wmma_bf16(ak[0], bq0, zz);
        zz = wmma_bf16(ak[1], bq1, zz);
        c[nt] = zz;
    }

    // ---- (optional) causal mask + block row-max ----
    float bm[8];
#pragma unroll
    for (int r = 0; r < 8; ++r) bm[r] = -1e30f;
#pragma unroll
    for (int nt = 0; nt < 2; ++nt) {
        const int s = s0 + nt * 16 + cN;
#pragma unroll
        for (int r = 0; r < 8; ++r) {
            float v = c[nt][r];
            if (MASK) {
                const int t = t0 + ch * 8 + r;
                if (s > t) v = -1e30f;
                c[nt][r] = v;
            }
            bm[r] = fmaxf(bm[r], v);
        }
    }
#pragma unroll
    for (int d = 1; d < 16; d <<= 1)
#pragma unroll
        for (int r = 0; r < 8; ++r)
            bm[r] = fmaxf(bm[r], __shfl_xor(bm[r], d, 32));

    // ---- online softmax rescale ----
#pragma unroll
    for (int r = 0; r < 8; ++r) {
        const float mn = fmaxf(mrun[r], bm[r]);
        const float sc = __expf(mrun[r] - mn);
        mrun[r] = mn;
        lrun[r] *= sc;
#pragma unroll
        for (int dt = 0; dt < 4; ++dt) acc[dt][r] *= sc;
    }

    // ---- P = exp(S - m), row sums ----
    float rs[8];
#pragma unroll
    for (int r = 0; r < 8; ++r) rs[r] = 0.0f;
#pragma unroll
    for (int nt = 0; nt < 2; ++nt)
#pragma unroll
        for (int r = 0; r < 8; ++r) {
            const float p = __expf(c[nt][r] - mrun[r]);
            c[nt][r] = p;
            rs[r] += p;
        }
#pragma unroll
    for (int d = 1; d < 16; d <<= 1)
#pragma unroll
        for (int r = 0; r < 8; ++r) rs[r] += __shfl_xor(rs[r], d, 32);
#pragma unroll
    for (int r = 0; r < 8; ++r) lrun[r] += rs[r];

    // ---- re-shape P (C-layout f32) -> A-fragment bf16 via LDS ----
    __syncthreads();
#pragma unroll
    for (int nt = 0; nt < 2; ++nt)
#pragma unroll
        for (int r = 0; r < 8; ++r)
            pl[(ch * 8 + r) * 32 + nt * 16 + cN] = f2bf(c[nt][r]);
    __syncthreads();
    Frag pa;
    {
        const uint4* lp = (const uint4*)(pl + cN * 32 + ch * 8);
        pa.u[0] = lp[0];
        pa.u[1] = lp[2];
    }

    // ---- O += P @ V  (V consumed transposed: contiguous b128 loads) ----
#pragma unroll
    for (int dt = 0; dt < 4; ++dt) {
        Frag bv;
        const uint4* vp = (const uint4*)(vt + (size_t)(dt * 16 + cN) * T_LEN
                                            + s0 + ch * 8);
        bv.u[0] = vp[0];
        bv.u[1] = vp[2];
        acc[dt] = wmma_bf16(pa, bv, acc[dt]);
    }
}

// Kb/Qb: [H][B*T][H_D] bf16; VT: [H][B][H_D][T_LEN] bf16; Out: [B*T][D] bf16.
__global__ __launch_bounds__(32)
void k_attention(const unsigned short* __restrict__ Kb,
                 const unsigned short* __restrict__ Qb,
                 const unsigned short* __restrict__ VT,
                 unsigned short* __restrict__ Out) {
    __shared__ __align__(16) unsigned short pl[16 * 32];
    const int lane = threadIdx.x;
    const int cN   = lane & 15;
    const int ch   = lane >> 4;
    const int t0   = blockIdx.x * 16;
    const int bh   = blockIdx.y;
    const int b    = bh / N_H;
    const int hh   = bh % N_H;

    const unsigned short* kh = Kb + ((size_t)hh * B_T + (size_t)b * T_LEN) * H_D;
    const unsigned short* qh = Qb + ((size_t)hh * B_T + (size_t)b * T_LEN) * H_D;
    const unsigned short* vt = VT + ((size_t)hh * N_B + b) * (size_t)H_D * T_LEN;

    // K-row fragments (A operand), invariant over the s loop. d = 0..63 -> 2 frags.
    Frag ak[2];
#pragma unroll
    for (int ks = 0; ks < 2; ++ks) {
        const uint4* p = (const uint4*)(kh + (size_t)(t0 + cN) * H_D + ks * 32 + ch * 8);
        ak[ks].u[0] = p[0];
        ak[ks].u[1] = p[2];
    }

    float mrun[8], lrun[8];
    f32x8 acc[4];
#pragma unroll
    for (int r = 0; r < 8; ++r) { mrun[r] = -1e30f; lrun[r] = 0.0f; }
#pragma unroll
    for (int dt = 0; dt < 4; ++dt) acc[dt] = (f32x8){};

    // Phase 1: fully-valid blocks (no masking on the hot path).
    int s0 = 0;
    for (; s0 + 31 <= t0; s0 += 32)
        attn_block<false>(s0, t0, cN, ch, qh, vt, ak, pl, mrun, lrun, acc);
    // Phase 2: diagonal block(s) with causal masking (at most one).
    for (; s0 <= t0 + 15; s0 += 32)
        attn_block<true>(s0, t0, cN, ch, qh, vt, ak, pl, mrun, lrun, acc);

    // ---- normalize + store (concat-head layout [B*T][D]) ----
#pragma unroll
    for (int dt = 0; dt < 4; ++dt)
#pragma unroll
        for (int r = 0; r < 8; ++r) {
            const float o = acc[dt][r] / lrun[r];
            Out[(size_t)(b * T_LEN + t0 + ch * 8 + r) * D_DIM
                + hh * H_D + dt * 16 + cN] = f2bf(o);
        }
}

// ---------------------------------------------------------------------- launch
extern "C" void kernel_launch(void* const* d_in, const int* in_sizes, int n_in,
                              void* d_out, int out_size, void* d_ws, size_t ws_size,
                              hipStream_t stream) {
    (void)in_sizes; (void)n_in; (void)out_size; (void)ws_size;
    const float* x  = (const float*)d_in[0];
    const float* Wi = (const float*)d_in[1];
    const float* bi = (const float*)d_in[2];
    const float* Wk = (const float*)d_in[3];
    const float* bk = (const float*)d_in[4];
    const float* Wq = (const float*)d_in[5];
    const float* bq = (const float*)d_in[6];
    const float* Wv = (const float*)d_in[7];
    const float* bv = (const float*)d_in[8];
    const float* Wo = (const float*)d_in[9];
    const float* bo = (const float*)d_in[10];
    float* out = (float*)d_out;

    char* ws = (char*)d_ws;
    size_t off = 0;
    auto take = [&](size_t elems) {
        unsigned short* p = (unsigned short*)(ws + off);
        off += ((elems * 2) + 255) & ~(size_t)255;
        return p;
    };
    unsigned short* xb   = take((size_t)B_T * D_DIM);        // x bf16
    unsigned short* WiT  = take((size_t)D_DIM * D_DIM);      // Wi^T bf16
    unsigned short* WkT  = take((size_t)N_H * H_D * D_DIM);
    unsigned short* WqT  = take((size_t)N_H * H_D * D_DIM);
    unsigned short* WvT  = take((size_t)N_H * H_D * D_DIM);
    unsigned short* WoT  = take((size_t)D_DIM * D_DIM);
    unsigned short* hb   = take((size_t)B_T * D_DIM);        // h bf16
    unsigned short* kb   = take((size_t)N_H * B_T * H_D);    // [H][B*T][HD]
    unsigned short* qb   = take((size_t)N_H * B_T * H_D);
    unsigned short* vtb  = take((size_t)N_H * N_B * H_D * T_LEN);  // [H][B][HD][T]
    unsigned short* attb = take((size_t)B_T * D_DIM);        // concat-head bf16

    // fp32 -> bf16 conversions (weights stored transposed for WMMA B-frags)
    {
        int n = B_T * D_DIM;
        k_f32_to_bf16<<<(n + 255) / 256, 256, 0, stream>>>(x, xb, n);
    }
    {
        int tot = D_DIM * D_DIM;
        k_f32_to_bf16_T<<<(tot + 255) / 256, 256, 0, stream>>>(Wi, WiT, D_DIM, D_DIM, tot);
        k_f32_to_bf16_T<<<(tot + 255) / 256, 256, 0, stream>>>(Wo, WoT, D_DIM, D_DIM, tot);
    }
    {
        int tot = N_H * D_DIM * H_D;
        k_f32_to_bf16_T<<<(tot + 255) / 256, 256, 0, stream>>>(Wk, WkT, D_DIM, H_D, tot);
        k_f32_to_bf16_T<<<(tot + 255) / 256, 256, 0, stream>>>(Wq, WqT, D_DIM, H_D, tot);
        k_f32_to_bf16_T<<<(tot + 255) / 256, 256, 0, stream>>>(Wv, WvT, D_DIM, H_D, tot);
    }

    const dim3 gblk(128, 1, 1);  // 4 waves per block (128x64 tile)

    // h = x @ Wi + bi
    k_gemm_wmma<0><<<dim3(D_DIM / 64, B_T / 128, 1), gblk, 0, stream>>>(
        xb, WiT, bi, hb, B_T, D_DIM, D_DIM, 0, 0, 0, 0);

    // per-head projections (batched over grid.z = head)
    k_gemm_wmma<0><<<dim3(H_D / 64, B_T / 128, N_H), gblk, 0, stream>>>(
        hb, WkT, bk, kb, B_T, H_D, D_DIM,
        0, (long long)H_D * D_DIM, H_D, (long long)B_T * H_D);
    k_gemm_wmma<0><<<dim3(H_D / 64, B_T / 128, N_H), gblk, 0, stream>>>(
        hb, WqT, bq, qb, B_T, H_D, D_DIM,
        0, (long long)H_D * D_DIM, H_D, (long long)B_T * H_D);
    // v stored transposed ([H][B][HD][T]) for the PV matmul
    k_gemm_wmma<1><<<dim3(H_D / 64, B_T / 128, N_H), gblk, 0, stream>>>(
        hb, WvT, bv, vtb, B_T, H_D, D_DIM,
        0, (long long)H_D * D_DIM, H_D, (long long)N_B * H_D * T_LEN);

    // flash-style attention (one wave per block)
    k_attention<<<dim3(T_LEN / 16, N_B * N_H, 1), dim3(32, 1, 1), 0, stream>>>(
        kb, qb, vtb, attb);

    // out = att @ Wo + bo  (fp32 output)
    k_gemm_wmma<2><<<dim3(D_DIM / 64, B_T / 128, 1), gblk, 0, stream>>>(
        attb, WoT, bo, out, B_T, D_DIM, D_DIM, 0, 0, 0, 0);
}
